// PrototypeGuidedTaskDecoder_81441169867237
// MI455X (gfx1250) — matrix operand order repaired and checked by reading, not compiled, not run on hardware
//
#include <hip/hip_runtime.h>
#include <hip/hip_bf16.h>
#include <math.h>

// ---------------------------------------------------------------------------
// PrototypeGuidedTaskDecoder for MI455X (gfx1250, wave32).
// GEMM-shaped ops (1x1 convs) run on v_wmma_f32_16x16x32_f16 (f16 in, f32 acc),
// templated on K (fully unrolled) with a 16x32 output tile per wave
// (one A fragment, two B fragments, two accumulators).
// Attention is algebraically reduced (2 queries): scores = Acoef . sf, and
// att = wv . (sf @ attn), avoiding k/v materialization.
// Workspace requirement: ~120.2 MB (see offsets below).
// ---------------------------------------------------------------------------

typedef __attribute__((ext_vector_type(16))) _Float16 v16h;
typedef __attribute__((ext_vector_type(8)))  float    v8f;

#define DEVI __device__ __forceinline__

DEVI float gelu_exact(float x){ return 0.5f*x*(1.0f+erff(x*0.70710678118654752440f)); }
DEVI float sigmoidf_(float x){ return 1.0f/(1.0f+expf(-x)); }

static constexpr int BN  = 4;
static constexpr int NP  = 16384;   // 128*128 pixels

// workspace offsets (floats)
static constexpr size_t O_A  = 0;                       // 8,388,608  (xh_up -> sf)
static constexpr size_t O_B  = 8388608;                 // 12,582,912 (xcat -> sf1 -> h)
static constexpr size_t O_X  = 20971520;                // 8,388,608  (x)
static constexpr size_t O_FS = 29360128;                // 65,536 fused_sim
static constexpr size_t O_SC = 29425664;                // 524,288 scores/prob
static constexpr size_t O_LG = 29949952;                // 65,536 logit
static constexpr size_t O_SM = 30015488;                // small region
// small region sub-offsets (floats, relative to O_SM)
static constexpr size_t ST_PL=0, ST_PH=8, ST_FU=16, ST_SE=24, ST_HD=56, SUMABS=64;
static constexpr size_t MR_PL=72, MR_PH=80, MR_FU=88, MR_SE=96, MR_HD=128;
static constexpr size_t S_FG=256, S_BG=768, S_PINIT=1280, S_Q=2304, S_AC=3328,
                        S_KC=7424, S_T=7456, S_A1=11552, S_A2=12576, S_PF=13600,
                        S_EKH=14112, S_EW=15136, S_WEFF=15168; // end 19776

// output offsets (floats)
static constexpr size_t O_PRED=0, O_Y=1048576, O_FSW=9437184;

// ---------------------------------------------------------------------------
// WMMA 1x1-conv GEMM: Out[b][m][n] = sum_k W[m][k]*In[b][k][n] (+bias)
// One wave computes a 16(M) x 32(N) tile; K fully unrolled (template), f16
// fragments, f32 accumulate. Optional per-batch (groups=1) GN stats.
// ---------------------------------------------------------------------------
template<int K, bool HAS_BIAS, bool HAS_STATS>
__global__ void __launch_bounds__(256)
conv1x1_wmma_kernel(const float* __restrict__ in, const float* __restrict__ w,
                    const float* __restrict__ bias, float* __restrict__ out,
                    size_t in_bstride, size_t out_bstride,
                    int mt_n, float* __restrict__ stats)
{
    constexpr int NT2 = NP / 32;      // 512 double-wide N tiles
    int wave = blockIdx.x * 8 + (threadIdx.x >> 5);
    int l  = threadIdx.x & 31;
    int lm = l & 15;
    int hi = l >> 4;                  // 0 for lanes 0-15, 1 for lanes 16-31
    int tiles_per_b = mt_n * NT2;
    int b  = wave / tiles_per_b;
    int r  = wave % tiles_per_b;
    int mt = r / NT2, nt = r % NT2;
    int m0 = mt * 16;
    int n  = nt * 32 + lm;

    const float* inb = in + (size_t)b * in_bstride;
    const float* wr  = w + (size_t)(m0 + lm) * K;
    const int kbA = hi * 16;          // A: lanes>=16 hold K+16 group
    const int kbB = hi * 16;          // B: lanes>=16 hold K+16 group

    v8f acc0 = {}, acc1 = {};
    #pragma unroll
    for (int k0 = 0; k0 < K; k0 += 32) {
        v16h a, b0, b1;
        const float* wp = wr + k0 + kbA;
        #pragma unroll
        for (int v = 0; v < 4; ++v) {        // A VGPR0..3: K = kbA+2v, +1
            a[2*v]   = (_Float16)wp[2*v];
            a[2*v+1] = (_Float16)wp[2*v+1];
            a[8+2*v] = (_Float16)wp[8+2*v];   // A VGPR4..7: K = 8+kbA+2v, +1
            a[9+2*v] = (_Float16)wp[8+2*v+1];
        }
        const float* ip = inb + (size_t)(k0 + kbB) * NP + n;
        #pragma unroll
        for (int v = 0; v < 8; ++v) {        // B: V_v holds rows K=kbB+2v, +1
            b0[2*v]   = (_Float16)ip[(size_t)(2*v)   * NP];
            b0[2*v+1] = (_Float16)ip[(size_t)(2*v+1) * NP];
            b1[2*v]   = (_Float16)ip[(size_t)(2*v)   * NP + 16];
            b1[2*v+1] = (_Float16)ip[(size_t)(2*v+1) * NP + 16];
        }
        acc0 = __builtin_amdgcn_wmma_f32_16x16x32_f16(
                   false, a, false, b0, (short)0, acc0, false, false);
        acc1 = __builtin_amdgcn_wmma_f32_16x16x32_f16(
                   false, a, false, b1, (short)0, acc1, false, false);
    }

    float s = 0.f, s2 = 0.f;
    float* ob = out + (size_t)b * out_bstride + n;
    #pragma unroll
    for (int v = 0; v < 8; ++v) {            // D: VGPR v -> M = m0+v+8*hi
        int m = m0 + v + 8 * hi;
        float bv = HAS_BIAS ? bias[m] : 0.f;
        float v0 = acc0[v] + bv;
        float v1 = acc1[v] + bv;
        ob[(size_t)m * NP]      = v0;
        ob[(size_t)m * NP + 16] = v1;
        if (HAS_STATS) { s += v0 + v1; s2 += v0*v0 + v1*v1; }
    }
    if (HAS_STATS) {
        #pragma unroll
        for (int off = 16; off > 0; off >>= 1) {
            s  += __shfl_down(s,  off, 32);
            s2 += __shfl_down(s2, off, 32);
        }
        if (l == 0) {
            atomicAdd(&stats[b*2],   s);
            atomicAdd(&stats[b*2+1], s2);
        }
    }
}

// ---------------------------------------------------------------------------
__global__ void resize_ac_kernel(const float* __restrict__ src, float* __restrict__ dst,
                                 int Bc, int IH, int IW, int OH, int OW)
{
    size_t idx = (size_t)blockIdx.x * 256 + threadIdx.x;
    size_t total = (size_t)Bc * OH * OW;
    if (idx >= total) return;
    int ox = (int)(idx % OW);
    size_t t = idx / OW;
    int oy = (int)(t % OH);
    int bc = (int)(t / OH);
    float sy = (OH > 1) ? oy * (float)(IH - 1) / (float)(OH - 1) : 0.f;
    float sx = (OW > 1) ? ox * (float)(IW - 1) / (float)(OW - 1) : 0.f;
    int y0 = (int)floorf(sy); int y1 = min(y0 + 1, IH - 1); float fy = sy - y0;
    int x0 = (int)floorf(sx); int x1 = min(x0 + 1, IW - 1); float fx = sx - x0;
    const float* s = src + (size_t)bc * IH * IW;
    float v00 = s[y0*IW + x0], v01 = s[y0*IW + x1];
    float v10 = s[y1*IW + x0], v11 = s[y1*IW + x1];
    float t0 = v00 * (1.f - fy) + v10 * fy;
    float t1 = v01 * (1.f - fy) + v11 * fy;
    dst[idx] = t0 * (1.f - fx) + t1 * fx;
}

__global__ void finalize_stats_kernel(const float* __restrict__ st, float* __restrict__ mr,
                                      int n, float invc)
{
    int t = threadIdx.x;
    if (t >= n) return;
    float s = st[2*t], s2 = st[2*t+1];
    float mean = s * invc;
    float var  = s2 * invc - mean * mean;
    mr[2*t]   = mean;
    mr[2*t+1] = rsqrtf(var + 1e-5f);
}

__global__ void gn_act_kernel(float* __restrict__ data, size_t bstride, int C,
                              const float* __restrict__ g, const float* __restrict__ be,
                              const float* __restrict__ mr, int groups, int do_gelu)
{
    size_t idx = (size_t)blockIdx.x * 256 + threadIdx.x;
    int p = (int)(idx % NP);
    size_t t = idx / NP;
    int c = (int)(t % C);
    int b = (int)(t / C);
    int gi = c / (C / groups);
    float mean = mr[(b*groups + gi)*2];
    float rstd = mr[(b*groups + gi)*2 + 1];
    float* ptr = data + (size_t)b * bstride + (size_t)c * NP + p;
    float v = (*ptr - mean) * rstd * g[c] + be[c];
    if (do_gelu) v = gelu_exact(v);
    *ptr = v;
}

// grouped 3x3 conv (groups=32, 4in->4out per group) + GN(groups=4) stats
__global__ void __launch_bounds__(256)
se_conv3x3_kernel(const float* __restrict__ x, const float* __restrict__ w,
                  const float* __restrict__ bias, float* __restrict__ out,
                  float* __restrict__ stats)
{
    __shared__ float sb[256];
    int p = blockIdx.x * 256 + threadIdx.x;
    int oc = blockIdx.y, b = blockIdx.z;
    int y0 = p >> 7, x0 = p & 127;
    int gbase = oc & ~3;
    float acc = bias[oc];
    for (int i = 0; i < 4; ++i) {
        const float* xp = x + (size_t)(b*128 + gbase + i) * NP;
        const float* wp = w + (oc*4 + i) * 9;
        #pragma unroll
        for (int ky = 0; ky < 3; ++ky) {
            int yy = y0 + ky - 1; if (yy < 0 || yy > 127) continue;
            #pragma unroll
            for (int kx = 0; kx < 3; ++kx) {
                int xx = x0 + kx - 1; if (xx < 0 || xx > 127) continue;
                acc += wp[ky*3+kx] * xp[yy*128 + xx];
            }
        }
    }
    out[(size_t)(b*128 + oc) * NP + p] = acc;
    float s = acc;
    sb[threadIdx.x] = s; __syncthreads();
    for (int st = 128; st > 0; st >>= 1) { if (threadIdx.x < st) sb[threadIdx.x] += sb[threadIdx.x+st]; __syncthreads(); }
    float rs = sb[0]; __syncthreads();
    sb[threadIdx.x] = s * s; __syncthreads();
    for (int st = 128; st > 0; st >>= 1) { if (threadIdx.x < st) sb[threadIdx.x] += sb[threadIdx.x+st]; __syncthreads(); }
    if (threadIdx.x == 0) {
        int g = b*4 + (oc >> 5);
        atomicAdd(&stats[g*2],   rs);
        atomicAdd(&stats[g*2+1], sb[0]);
    }
}

// depthwise 3x3 + per-batch GN stats
__global__ void __launch_bounds__(256)
dwconv3x3_kernel(const float* __restrict__ x, const float* __restrict__ w,
                 const float* __restrict__ bias, float* __restrict__ out,
                 float* __restrict__ stats)
{
    __shared__ float sb[256];
    int p = blockIdx.x * 256 + threadIdx.x;
    int c = blockIdx.y, b = blockIdx.z;
    int y0 = p >> 7, x0 = p & 127;
    const float* xp = x + (size_t)(b*128 + c) * NP;
    const float* wp = w + c * 9;
    float acc = bias[c];
    #pragma unroll
    for (int ky = 0; ky < 3; ++ky) {
        int yy = y0 + ky - 1; if (yy < 0 || yy > 127) continue;
        #pragma unroll
        for (int kx = 0; kx < 3; ++kx) {
            int xx = x0 + kx - 1; if (xx < 0 || xx > 127) continue;
            acc += wp[ky*3+kx] * xp[yy*128 + xx];
        }
    }
    out[(size_t)(b*128 + c) * NP + p] = acc;
    sb[threadIdx.x] = acc; __syncthreads();
    for (int st = 128; st > 0; st >>= 1) { if (threadIdx.x < st) sb[threadIdx.x] += sb[threadIdx.x+st]; __syncthreads(); }
    float rs = sb[0]; __syncthreads();
    sb[threadIdx.x] = acc * acc; __syncthreads();
    for (int st = 128; st > 0; st >>= 1) { if (threadIdx.x < st) sb[threadIdx.x] += sb[threadIdx.x+st]; __syncthreads(); }
    if (threadIdx.x == 0) {
        atomicAdd(&stats[b*2],   rs);
        atomicAdd(&stats[b*2+1], sb[0]);
    }
}

// prototypes: pinit MLP + row-normalize + table lookup -> fg/bg  (<<<1,256>>>)
__global__ void protos_kernel(const float* __restrict__ emb, const float* __restrict__ w1,
                              const float* __restrict__ b1, const float* __restrict__ w2,
                              const float* __restrict__ b2, const int* __restrict__ task_ids,
                              float* __restrict__ pinit, float* __restrict__ fg, float* __restrict__ bg)
{
    __shared__ float hid[BN][256];
    __shared__ float nrm[8];
    int t = threadIdx.x;
    for (int b = 0; b < BN; ++b) {
        float a = b1[t];
        for (int j = 0; j < 64; ++j) a += w1[t*64 + j] * emb[b*64 + j];
        hid[b][t] = gelu_exact(a);
    }
    __syncthreads();
    for (int b = 0; b < BN; ++b) {
        float a = b2[t];
        for (int j = 0; j < 256; ++j) a += w2[t*256 + j] * hid[b][j];
        pinit[b*256 + t] = a;
    }
    __syncthreads();
    if (t < 8) {
        float s = 0.f;
        for (int c = 0; c < 128; ++c) { float v = pinit[t*128 + c]; s += v*v; }
        nrm[t] = fmaxf(sqrtf(s), 1e-12f);
    }
    __syncthreads();
    for (int i = t; i < 1024; i += 256) pinit[i] /= nrm[i >> 7];
    __syncthreads();
    for (int i = t; i < 512; i += 256) {
        int b = i >> 7, c = i & 127;
        int tid_b = task_ids[b], bp = b;
        for (int j = 0; j < BN; ++j) { if (task_ids[j] == tid_b) { bp = j; break; } }
        fg[b*128 + c] = pinit[(bp*2 + 0)*128 + c];
        bg[b*128 + c] = pinit[(bp*2 + 1)*128 + c];
    }
}

// fused similarity map + global sum|fs|
__global__ void __launch_bounds__(256)
fsim_kernel(const float* __restrict__ sf, const float* __restrict__ fg,
            const float* __restrict__ bg, float* __restrict__ fs, float* __restrict__ sumabs)
{
    __shared__ float sb[256];
    int p = blockIdx.x * 256 + threadIdx.x;
    int b = blockIdx.y;
    const float* base = sf + (size_t)b * 128 * NP + p;
    float s2 = 0.f, df = 0.f, db = 0.f;
    for (int c = 0; c < 128; ++c) {
        float v = base[(size_t)c * NP];
        s2 += v*v; df += v*fg[b*128 + c]; db += v*bg[b*128 + c];
    }
    float val = (df - db) / fmaxf(sqrtf(s2), 1e-12f);
    fs[(size_t)b * NP + p] = val;
    sb[threadIdx.x] = fabsf(val); __syncthreads();
    for (int st = 128; st > 0; st >>= 1) { if (threadIdx.x < st) sb[threadIdx.x] += sb[threadIdx.x+st]; __syncthreads(); }
    if (threadIdx.x == 0) atomicAdd(sumabs, sb[0]);
}

// q = wq @ pq + bq      (1024 threads)
__global__ void q_kernel(const float* __restrict__ fg, const float* __restrict__ bg,
                         const float* __restrict__ at_in_w, const float* __restrict__ at_in_b,
                         float* __restrict__ q)
{
    int idx = blockIdx.x * 256 + threadIdx.x;
    int b = idx >> 8, qi = (idx >> 7) & 1, oc = idx & 127;
    const float* pq = (qi ? bg : fg) + b*128;
    float s = at_in_b[oc];
    for (int c = 0; c < 128; ++c) s += at_in_w[oc*128 + c] * pq[c];
    q[idx] = s;
}

// Acoef[b,qi,h,c] = sum_d q[b,qi,h*32+d]*wk[h*32+d][c];  kconst = q . bk
__global__ void acoef_kernel(const float* __restrict__ q, const float* __restrict__ at_in_w,
                             const float* __restrict__ at_in_b, float* __restrict__ ac,
                             float* __restrict__ kc)
{
    int idx = blockIdx.x * 256 + threadIdx.x;   // 4096
    int b = idx >> 10, qi = (idx >> 9) & 1, h = (idx >> 7) & 3, c = idx & 127;
    const float* qv = q + (b*2 + qi)*128 + h*32;
    float s = 0.f;
    for (int d = 0; d < 32; ++d) s += qv[d] * at_in_w[(128 + h*32 + d)*128 + c];
    ac[idx] = s;
    if (c == 0) {
        float kcs = 0.f;
        for (int d = 0; d < 32; ++d) kcs += qv[d] * at_in_b[128 + h*32 + d];
        kc[(b*2 + qi)*4 + h] = kcs;
    }
}

__global__ void scores_kernel(const float* __restrict__ sf, const float* __restrict__ ac,
                              const float* __restrict__ kc, float* __restrict__ sc)
{
    int p = blockIdx.x * 256 + threadIdx.x;
    int r = blockIdx.y;            // 32 rows: ((b*2+qi)*4+h)
    int b = r >> 3;
    const float* a = ac + r*128;
    const float* base = sf + (size_t)b * 128 * NP + p;
    float s = kc[r];
    for (int c = 0; c < 128; ++c) s += a[c] * base[(size_t)c * NP];
    sc[(size_t)r * NP + p] = s * 0.17677669529663687f;   // 1/sqrt(32)
}

__global__ void __launch_bounds__(256)
softmax_row_kernel(float* __restrict__ sc)
{
    __shared__ float sb[256];
    float* row = sc + (size_t)blockIdx.x * NP;
    int t = threadIdx.x;
    float m = -3.4e38f;
    for (int p = t; p < NP; p += 256) m = fmaxf(m, row[p]);
    sb[t] = m; __syncthreads();
    for (int st = 128; st > 0; st >>= 1) { if (t < st) sb[t] = fmaxf(sb[t], sb[t+st]); __syncthreads(); }
    m = sb[0]; __syncthreads();
    float s = 0.f;
    for (int p = t; p < NP; p += 256) s += expf(row[p] - m);
    sb[t] = s; __syncthreads();
    for (int st = 128; st > 0; st >>= 1) { if (t < st) sb[t] += sb[t+st]; __syncthreads(); }
    float inv = 1.f / sb[0]; __syncthreads();
    for (int p = t; p < NP; p += 256) row[p] = expf(row[p] - m) * inv;
}

__global__ void __launch_bounds__(256)
attnT_kernel(const float* __restrict__ prob, const float* __restrict__ sf, float* __restrict__ T)
{
    __shared__ float sb[256];
    int bid = blockIdx.x;          // 4096
    int r = bid >> 7, c = bid & 127;
    int b = r >> 3;
    const float* pr = prob + (size_t)r * NP;
    const float* sp = sf + (size_t)(b*128 + c) * NP;
    float s = 0.f;
    for (int p = threadIdx.x; p < NP; p += 256) s += pr[p] * sp[p];
    sb[threadIdx.x] = s; __syncthreads();
    for (int st = 128; st > 0; st >>= 1) { if (threadIdx.x < st) sb[threadIdx.x] += sb[threadIdx.x+st]; __syncthreads(); }
    if (threadIdx.x == 0) T[r*128 + c] = sb[0];
}

// att1 = wv . T + bv   (per head block of T)
__global__ void att1_kernel(const float* __restrict__ T, const float* __restrict__ at_in_w,
                            const float* __restrict__ at_in_b, float* __restrict__ att1)
{
    int idx = blockIdx.x * 256 + threadIdx.x;  // 1024
    int b = idx >> 8, qi = (idx >> 7) & 1, oc = idx & 127;
    int h = oc >> 5;
    const float* Tv = T + ((b*2 + qi)*4 + h)*128;
    float s = at_in_b[256 + oc];
    for (int c = 0; c < 128; ++c) s += at_in_w[(256 + oc)*128 + c] * Tv[c];
    att1[idx] = s;
}

__global__ void att2_kernel(const float* __restrict__ att1, const float* __restrict__ w,
                            const float* __restrict__ be, float* __restrict__ att2)
{
    int idx = blockIdx.x * 256 + threadIdx.x;  // 1024
    int b = idx >> 8, qi = (idx >> 7) & 1, oc = idx & 127;
    const float* av = att1 + (b*2 + qi)*128;
    float s = be[oc];
    for (int c = 0; c < 128; ++c) s += w[oc*128 + c] * av[c];
    att2[idx] = s;
}

__global__ void protofeat_kernel(const float* __restrict__ att2, const float* __restrict__ w,
                                 const float* __restrict__ be, float* __restrict__ pf)
{
    int idx = blockIdx.x * 256 + threadIdx.x;  // 512
    int b = idx >> 7, oc = idx & 127;
    const float* fga = att2 + (b*2 + 0)*128;
    const float* bga = att2 + (b*2 + 1)*128;
    float s = be[oc];
    const float* wr = w + (size_t)oc * 384;
    for (int c = 0; c < 128; ++c) {
        s += wr[c] * fga[c];
        s += wr[128 + c] * bga[c];
        s += wr[256 + c] * (fga[c] - bga[c]);
    }
    pf[b*128 + oc] = s;
}

__global__ void ekh_kernel(const float* __restrict__ pf, const float* __restrict__ w1,
                           const float* __restrict__ b1, float* __restrict__ ekh)
{
    int idx = blockIdx.x * 256 + threadIdx.x;  // 1024
    int b = idx >> 8, j = idx & 255;
    float s = b1[j];
    for (int c = 0; c < 128; ++c) s += w1[j*128 + c] * pf[b*128 + c];
    ekh[idx] = gelu_exact(s);
}

__global__ void ew_kernel(const float* __restrict__ pf, const float* __restrict__ w,
                          const float* __restrict__ be, float* __restrict__ ew)
{
    __shared__ float sl[6];
    int b = blockIdx.x, t = threadIdx.x;
    if (t < 6) {
        float s = be[t];
        for (int c = 0; c < 128; ++c) s += w[t*128 + c] * pf[b*128 + c];
        sl[t] = s;
    }
    __syncthreads();
    if (t < 6) {
        float m = sl[0];
        for (int i = 1; i < 6; ++i) m = fmaxf(m, sl[i]);
        float ssum = 0.f;
        for (int i = 0; i < 6; ++i) ssum += expf(sl[i] - m);
        ew[b*6 + t] = expf(sl[t] - m) / ssum;
    }
}

__global__ void weff_kernel(const float* __restrict__ ekh, const float* __restrict__ w2,
                            const float* __restrict__ b2, const float* __restrict__ ew,
                            const float* __restrict__ eis, float* __restrict__ weff)
{
    int idx = blockIdx.x * 256 + threadIdx.x;
    if (idx >= 4608) return;                    // (b*128+c)*9+r
    int b = idx / 1152, rem = idx % 1152, c = rem / 9, r = rem % 9;
    float acc = 0.f;
    for (int m = 0; m < 6; ++m) {
        int row = (m*128 + c)*9 + r;
        const float* wr = w2 + (size_t)row * 256;
        float ekv = b2[row];
        for (int j = 0; j < 256; ++j) ekv += wr[j] * ekh[b*256 + j];
        acc += ew[b*6 + m] * ekv;
    }
    float e = eis[0];
    weff[idx] = e * acc + (1.f - e) * ((r == 4) ? 1.f : 0.f);
}

__global__ void dynconv_kernel(const float* __restrict__ x, const float* __restrict__ fs,
                               const float* __restrict__ weff, float* __restrict__ y)
{
    int p = blockIdx.x * 256 + threadIdx.x;
    int c = blockIdx.y, b = blockIdx.z;
    int y0 = p >> 7, x0 = p & 127;
    const float* xp = x + (size_t)(b*128 + c) * NP;
    const float* fp = fs + (size_t)b * NP;
    const float* we = weff + (b*128 + c) * 9;
    float acc = 0.f;
    #pragma unroll
    for (int r = 0; r < 9; ++r) {
        int yy = y0 + r/3 - 1, xx = x0 + r%3 - 1;
        if (yy < 0 || yy > 127 || xx < 0 || xx > 127) continue;
        int pp = yy*128 + xx;
        acc += we[r] * (xp[pp] * (1.f + 0.5f * sigmoidf_(fp[pp])));
    }
    y[(size_t)(b*128 + c) * NP + p] = acc;
}

__global__ void head_logit_kernel(const float* __restrict__ hact, const float* __restrict__ w2,
                                  const float* __restrict__ b2, const int* __restrict__ task_ids,
                                  const float* __restrict__ fs, const float* __restrict__ sumabs,
                                  const float* __restrict__ sim_temp, const float* __restrict__ proto_amp,
                                  float* __restrict__ logit)
{
    int p = blockIdx.x * 256 + threadIdx.x;
    int b = blockIdx.y;
    int tt = task_ids[b];
    const float* base = hact + (size_t)b * 128 * NP + p;
    const float* wr = w2 + tt * 128;
    float s = b2[tt];
    for (int c = 0; c < 128; ++c) s += wr[c] * base[(size_t)c * NP];
    float mean = sumabs[0] * (1.f / 65536.f);
    float fv = fs[(size_t)b * NP + p];
    float sim = tanhf(fv / (mean + 1e-6f) / fmaxf(sim_temp[0], 1e-3f));
    logit[(size_t)b * NP + p] = s + 5.0f * proto_amp[0] * sim;
}

__global__ void fsw_kernel(float* __restrict__ out)
{
    if (threadIdx.x < BN) out[threadIdx.x] = 1.0f;   // softmax over 1 elem == 1
}

// ---------------------------------------------------------------------------
extern "C" void kernel_launch(void* const* d_in, const int* in_sizes, int n_in,
                              void* d_out, int out_size, void* d_ws, size_t ws_size,
                              hipStream_t stream) {
    (void)in_sizes; (void)n_in; (void)out_size; (void)ws_size;
    const float* x_low    = (const float*)d_in[0];
    const float* x_high   = (const float*)d_in[1];
    const float* task_emb = (const float*)d_in[2];
    const float* pl_w = (const float*)d_in[3];
    const float* pl_g = (const float*)d_in[4];
    const float* pl_b = (const float*)d_in[5];
    const float* ph_w = (const float*)d_in[6];
    const float* ph_g = (const float*)d_in[7];
    const float* ph_b = (const float*)d_in[8];
    const float* fu_w = (const float*)d_in[9];
    const float* fu_g = (const float*)d_in[10];
    const float* fu_b = (const float*)d_in[11];
    const float* pi_w1 = (const float*)d_in[12];
    const float* pi_b1 = (const float*)d_in[13];
    const float* pi_w2 = (const float*)d_in[14];
    const float* pi_b2 = (const float*)d_in[15];
    const float* se_w1 = (const float*)d_in[16];
    const float* se_b1 = (const float*)d_in[17];
    const float* se_g  = (const float*)d_in[18];
    const float* se_be = (const float*)d_in[19];
    const float* se_w2 = (const float*)d_in[20];
    const float* se_b2 = (const float*)d_in[21];
    const float* at_in_w  = (const float*)d_in[22];
    const float* at_in_b  = (const float*)d_in[23];
    const float* at_out_w = (const float*)d_in[24];
    const float* at_out_b = (const float*)d_in[25];
    const float* pf_w = (const float*)d_in[26];
    const float* pf_b = (const float*)d_in[27];
    const float* eg_w1 = (const float*)d_in[33];
    const float* eg_b1 = (const float*)d_in[34];
    const float* eg_w2 = (const float*)d_in[35];
    const float* eg_b2 = (const float*)d_in[36];
    const float* ewg_w = (const float*)d_in[37];
    const float* ewg_b = (const float*)d_in[38];
    const float* hd_w1 = (const float*)d_in[39];
    const float* hd_b1 = (const float*)d_in[40];
    const float* hd_g  = (const float*)d_in[41];
    const float* hd_be = (const float*)d_in[42];
    const float* hd_w2 = (const float*)d_in[43];
    const float* hd_b2 = (const float*)d_in[44];
    const float* eis       = (const float*)d_in[45];
    const float* sim_temp  = (const float*)d_in[46];
    const float* proto_amp = (const float*)d_in[47];
    const int*   task_ids  = (const int*)d_in[48];

    float* ws  = (float*)d_ws;
    float* out = (float*)d_out;
    float* SM  = ws + O_SM;

    // zero GN-stat + sum|fs| accumulators every call (ws persists across replays)
    hipMemsetAsync(SM, 0, 72 * sizeof(float), stream);

    // 1) resize x_high 64x64 -> 128x128 into region A
    resize_ac_kernel<<<(4*128*128*128)/256, 256, 0, stream>>>(x_high, ws + O_A, 4*128, 64, 64, 128, 128);

    // 2) pl conv (64->64) into xcat channels 128..191 ; ph conv (128->128) into xcat 0..127
    conv1x1_wmma_kernel<64, false, true><<<1024, 256, 0, stream>>>(
        x_low, pl_w, nullptr, ws + O_B + (size_t)128*NP,
        (size_t)64*NP, (size_t)192*NP, 4, SM + ST_PL);
    conv1x1_wmma_kernel<128, false, true><<<2048, 256, 0, stream>>>(
        ws + O_A, ph_w, nullptr, ws + O_B,
        (size_t)128*NP, (size_t)192*NP, 8, SM + ST_PH);
    finalize_stats_kernel<<<1, 64, 0, stream>>>(SM + ST_PL, SM + MR_PL, 4, 1.f/(64.f*NP));
    finalize_stats_kernel<<<1, 64, 0, stream>>>(SM + ST_PH, SM + MR_PH, 4, 1.f/(128.f*NP));
    gn_act_kernel<<<(4*64*NP)/256, 256, 0, stream>>>(ws + O_B + (size_t)128*NP, (size_t)192*NP, 64,
                                                     pl_g, pl_b, SM + MR_PL, 1, 1);
    gn_act_kernel<<<(4*128*NP)/256, 256, 0, stream>>>(ws + O_B, (size_t)192*NP, 128,
                                                      ph_g, ph_b, SM + MR_PH, 1, 1);

    // 3) fuse conv (192->128) -> x
    conv1x1_wmma_kernel<192, false, true><<<2048, 256, 0, stream>>>(
        ws + O_B, fu_w, nullptr, ws + O_X,
        (size_t)192*NP, (size_t)128*NP, 8, SM + ST_FU);
    finalize_stats_kernel<<<1, 64, 0, stream>>>(SM + ST_FU, SM + MR_FU, 4, 1.f/(128.f*NP));
    gn_act_kernel<<<(4*128*NP)/256, 256, 0, stream>>>(ws + O_X, (size_t)128*NP, 128,
                                                      fu_g, fu_b, SM + MR_FU, 1, 1);

    // 4) se path: grouped 3x3 -> GN(4) -> gelu -> 1x1 (+bias) -> sf (region A)
    se_conv3x3_kernel<<<dim3(NP/256, 128, 4), 256, 0, stream>>>(ws + O_X, se_w1, se_b1, ws + O_B, SM + ST_SE);
    finalize_stats_kernel<<<1, 64, 0, stream>>>(SM + ST_SE, SM + MR_SE, 16, 1.f/(32.f*NP));
    gn_act_kernel<<<(4*128*NP)/256, 256, 0, stream>>>(ws + O_B, (size_t)128*NP, 128,
                                                      se_g, se_be, SM + MR_SE, 4, 1);
    conv1x1_wmma_kernel<128, true, false><<<2048, 256, 0, stream>>>(
        ws + O_B, se_w2, se_b2, ws + O_A,
        (size_t)128*NP, (size_t)128*NP, 8, nullptr);

    // 5) prototypes -> fg/bg ; fused similarity + sum|fs|
    protos_kernel<<<1, 256, 0, stream>>>(task_emb, pi_w1, pi_b1, pi_w2, pi_b2, task_ids,
                                         SM + S_PINIT, SM + S_FG, SM + S_BG);
    fsim_kernel<<<dim3(NP/256, 4), 256, 0, stream>>>(ws + O_A, SM + S_FG, SM + S_BG,
                                                     ws + O_FS, SM + SUMABS);

    // 6) attention (reduced form)
    q_kernel<<<4, 256, 0, stream>>>(SM + S_FG, SM + S_BG, at_in_w, at_in_b, SM + S_Q);
    acoef_kernel<<<16, 256, 0, stream>>>(SM + S_Q, at_in_w, at_in_b, SM + S_AC, SM + S_KC);
    scores_kernel<<<dim3(NP/256, 32), 256, 0, stream>>>(ws + O_A, SM + S_AC, SM + S_KC, ws + O_SC);
    softmax_row_kernel<<<32, 256, 0, stream>>>(ws + O_SC);
    attnT_kernel<<<4096, 256, 0, stream>>>(ws + O_SC, ws + O_A, SM + S_T);
    att1_kernel<<<4, 256, 0, stream>>>(SM + S_T, at_in_w, at_in_b, SM + S_A1);
    att2_kernel<<<4, 256, 0, stream>>>(SM + S_A1, at_out_w, at_out_b, SM + S_A2);
    protofeat_kernel<<<2, 256, 0, stream>>>(SM + S_A2, pf_w, pf_b, SM + S_PF);

    // 7) dynamic kernel weights
    ekh_kernel<<<4, 256, 0, stream>>>(SM + S_PF, eg_w1, eg_b1, SM + S_EKH);
    ew_kernel<<<4, 32, 0, stream>>>(SM + S_PF, ewg_w, ewg_b, SM + S_EW);
    weff_kernel<<<18, 256, 0, stream>>>(SM + S_EKH, eg_w2, eg_b2, SM + S_EW, eis, SM + S_WEFF);

    // 8) dynamic conv -> y (output)
    dynconv_kernel<<<dim3(NP/256, 128, 4), 256, 0, stream>>>(ws + O_X, ws + O_FS, SM + S_WEFF, out + O_Y);

    // 9) head: depthwise 3x3 -> GN -> gelu -> 1x1(task) + proto term
    dwconv3x3_kernel<<<dim3(NP/256, 128, 4), 256, 0, stream>>>(out + O_Y, hd_w1, hd_b1, ws + O_B, SM + ST_HD);
    finalize_stats_kernel<<<1, 64, 0, stream>>>(SM + ST_HD, SM + MR_HD, 4, 1.f/(128.f*NP));
    gn_act_kernel<<<(4*128*NP)/256, 256, 0, stream>>>(ws + O_B, (size_t)128*NP, 128,
                                                      hd_g, hd_be, SM + MR_HD, 1, 1);
    head_logit_kernel<<<dim3(NP/256, 4), 256, 0, stream>>>(ws + O_B, hd_w2, hd_b2, task_ids,
                                                           ws + O_FS, SM + SUMABS, sim_temp,
                                                           proto_amp, ws + O_LG);

    // 10) pred = resize 128 -> 512 ; fsw = ones
    resize_ac_kernel<<<(4*512*512)/256, 256, 0, stream>>>(ws + O_LG, out + O_PRED, 4, 128, 128, 512, 512);
    fsw_kernel<<<1, 32, 0, stream>>>(out + O_FSW);
}
